// MaskAttention_54924041781330
// MI455X (gfx1250) — compile-verified, matrix operand order checked by reading, not run on hardware
//
#include <hip/hip_runtime.h>
#include <hip/hip_bf16.h>

// ---------------------------------------------------------------------------
// MaskAttention for MI455X (gfx1250, wave32, WMMA 16x16x32 f16->f32)
//
//   1. cvt_wqkv / cvt_wout : f32 weights -> transposed f16 (B-operand layout)
//   2. ln_kernel           : LayerNorm f32 -> f16 xn
//   3. qkv_gemm            : block-tiled GEMM (256x64/block, 32 rows/wave),
//                            async B staging into double-buffered LDS
//   4. attn_kernel         : flash-attention, async K staging, tile-skip mask
//   5. out_gemm            : same GEMM structure, +bias, f32 out
// ---------------------------------------------------------------------------

typedef __attribute__((ext_vector_type(16))) _Float16 v16h;
typedef __attribute__((ext_vector_type(8)))  _Float16 v8h;
typedef __attribute__((ext_vector_type(8)))  float    v8f;

#define B_    4
#define N_    2048
#define H_    8
#define DH_   64
#define DIM_  512
#define ROWS_ (B_ * N_)        // 8192
#define QKVC_ 1536             // 3 * INNER

// ---------------------------------------------------------------------------
// CDNA5 async memory->LDS copy (16B per lane) + ASYNCcnt wait.
// GLOBAL_LOAD_ASYNC_TO_LDS_B128: VDST = LDS byte address VGPR,
// VADDR = 64-bit global address (ISA 15.18.3 opcode 98, tracked by ASYNCcnt).
// Generic pointers to __shared__ carry the LDS offset in their low 32 bits.
// ---------------------------------------------------------------------------
__device__ __forceinline__ void async_b128(void* lds_dst, const void* gsrc) {
  asm volatile("global_load_async_to_lds_b128 %0, %1, off"
               :
               : "v"((unsigned int)(unsigned long long)lds_dst),
                 "v"((unsigned long long)gsrc)
               : "memory");
}
__device__ __forceinline__ void wait_async0() {
  asm volatile("s_wait_asynccnt 0x0" ::: "memory");
}

// ---------------------------------------------------------------------------
// Fragment loader for WMMA A (row-major M x ld) and WMMA B when stored
// transposed (N rows of length K). CDNA5 16-bit A layout (ISA 7.12.2):
// lanes 0-15 hold M=0..15, halves 0-7 = K hi*8+0..7, halves 8-15 = K 16+hi*8.
// Two contiguous 16-byte loads per lane.
// ---------------------------------------------------------------------------
__device__ __forceinline__ v16h ldfragAB(const _Float16* base, int ld) {
  const int lane = threadIdx.x & 31;
  const int r    = lane & 15;
  const int hi   = lane >> 4;
  const _Float16* p = base + (size_t)r * ld + hi * 8;
  v8h lo = *(const v8h*)(p);
  v8h up = *(const v8h*)(p + 16);
  v16h out;
#pragma unroll
  for (int i = 0; i < 8; ++i) { out[i] = lo[i]; out[i + 8] = up[i]; }
  return out;
}

__device__ __forceinline__ v8f wmma_f16(v16h a, v16h b, v8f c) {
  return __builtin_amdgcn_wmma_f32_16x16x32_f16(false, a, false, b,
                                                (short)0, c, false, false);
}

// ---------------------------------------------------------------------------
// Weight conversion + transpose.
// ---------------------------------------------------------------------------
__global__ void cvt_wqkv_kernel(const float* __restrict__ w,
                                _Float16* __restrict__ wT) {
  int i = blockIdx.x * 256 + threadIdx.x;
  if (i < DIM_ * QKVC_) {
    int k = i / QKVC_, c = i % QKVC_;
    wT[(size_t)c * DIM_ + k] = (_Float16)w[i];
  }
}

__global__ void cvt_wout_kernel(const float* __restrict__ w,
                                _Float16* __restrict__ wT) {
  int i = blockIdx.x * 256 + threadIdx.x;
  if (i < DIM_ * DIM_) {
    int k = i / DIM_, c = i % DIM_;
    wT[(size_t)c * DIM_ + k] = (_Float16)w[i];
  }
}

// ---------------------------------------------------------------------------
// LayerNorm over DIM=512; one 256-thread block per row, f16 output.
// ---------------------------------------------------------------------------
__global__ void ln_kernel(const float* __restrict__ x,
                          const float* __restrict__ gamma,
                          const float* __restrict__ beta,
                          _Float16* __restrict__ xn) {
  const int row = blockIdx.x;
  const int t   = threadIdx.x;
  const float* xr = x + (size_t)row * DIM_;
  float a = xr[t];
  float c = xr[t + 256];

  __shared__ float rs[256];
  __shared__ float rq[256];
  rs[t] = a + c;
  rq[t] = a * a + c * c;
  __syncthreads();
#pragma unroll
  for (int s = 128; s > 0; s >>= 1) {
    if (t < s) { rs[t] += rs[t + s]; rq[t] += rq[t + s]; }
    __syncthreads();
  }
  const float mu   = rs[0] * (1.0f / DIM_);
  const float var  = rq[0] * (1.0f / DIM_) - mu * mu;
  const float rstd = rsqrtf(var + 1e-5f);

  _Float16* xo = xn + (size_t)row * DIM_;
  xo[t]       = (_Float16)((a - mu) * rstd * gamma[t] + beta[t]);
  xo[t + 256] = (_Float16)((c - mu) * rstd * gamma[t + 256] + beta[t + 256]);
}

// ---------------------------------------------------------------------------
// QKV GEMM, block-tiled: 256 threads = 8 waves; block computes 256 rows x 64
// cols, each wave 32 rows (2 A-frags, 8 accumulators). Per 32-k step the
// 64x32 B tile is async-DMA'd into LDS once per block (double buffered) and
// every staged B fragment feeds two back-to-back WMMAs. A fragments are
// software-pipelined in registers.
// ---------------------------------------------------------------------------
__global__ __launch_bounds__(256)
void qkv_gemm_kernel(const _Float16* __restrict__ xn,
                     const _Float16* __restrict__ wT,
                     _Float16* __restrict__ qb,
                     _Float16* __restrict__ kb,
                     _Float16* __restrict__ vb) {
  __shared__ __align__(16) _Float16 sB[2][64 * 32];   // [buf][col][k]

  const int NB = QKVC_ / 64;                  // 24 col-blocks
  const int bm = blockIdx.x / NB;
  const int bn = blockIdx.x % NB;
  const int wv = threadIdx.x >> 5;
  const int row0 = bm * 256 + wv * 32;
  const int col0 = bn * 64;

  const int lane = threadIdx.x & 31;
  const int ln   = lane & 15;
  const int hi   = lane >> 4;

  // One 16B chunk per thread per stage: 64 cols x 32 k halves = 4KB.
  const int scol = threadIdx.x >> 2;
  const int sko  = (threadIdx.x & 3) * 8;
  const _Float16* bsrc = wT + (size_t)(col0 + scol) * DIM_ + sko;

  async_b128(&sB[0][scol * 32 + sko], bsrc);
  wait_async0();
  __syncthreads();

  const _Float16* arow = xn + (size_t)row0 * DIM_;
  v8f acc[2][4] = {{v8f{}, v8f{}, v8f{}, v8f{}}, {v8f{}, v8f{}, v8f{}, v8f{}}};
  v16h a0 = ldfragAB(arow, DIM_);
  v16h a1 = ldfragAB(arow + (size_t)16 * DIM_, DIM_);

  int buf = 0;
  for (int kk = 0; kk < DIM_; kk += 32, buf ^= 1) {
    if (kk + 32 < DIM_)
      async_b128(&sB[buf ^ 1][scol * 32 + sko], bsrc + kk + 32);

    v16h a0n = a0, a1n = a1;
    if (kk + 32 < DIM_) {
      a0n = ldfragAB(arow + kk + 32, DIM_);
      a1n = ldfragAB(arow + (size_t)16 * DIM_ + kk + 32, DIM_);
    }

#pragma unroll
    for (int j = 0; j < 4; ++j) {
      v16h bfr = ldfragAB(&sB[buf][(j * 16) * 32], 32);
      acc[0][j] = wmma_f16(a0, bfr, acc[0][j]);   // fragment reused twice:
      acc[1][j] = wmma_f16(a1, bfr, acc[1][j]);   // hides next ds_load latency
    }
    a0 = a0n;
    a1 = a1n;
    wait_async0();
    __syncthreads();
  }

  const int which = col0 >> 9;                 // 0=q 1=k 2=v (uniform)
  const int h     = (col0 & 511) >> 6;         // head (uniform)
  _Float16* dst = (which == 0) ? qb : (which == 1) ? kb : vb;
  const int bidx = row0 >> 11;                 // batch (uniform per wave)
  const size_t base = ((size_t)(bidx * H_ + h)) * N_ * DH_;

#pragma unroll
  for (int m = 0; m < 2; ++m) {
#pragma unroll
    for (int j = 0; j < 4; ++j) {
#pragma unroll
      for (int r = 0; r < 8; ++r) {
        const int row = row0 + m * 16 + hi * 8 + r;
        const int n   = row & (N_ - 1);
        dst[base + (size_t)n * DH_ + j * 16 + ln] = (_Float16)acc[m][j][r];
      }
    }
  }
}

// ---------------------------------------------------------------------------
// Flash attention. Block = 128 threads (4 waves) owns 64 query rows of one
// (b,h); wave owns 16 rows. K tile (32x64) async-DMA'd to LDS; V tile stored
// transposed (needs register bounce). Modality segments are 512 tokens so the
// mask is uniform per 32-key tile / 64-query block: masked tiles are skipped.
// ---------------------------------------------------------------------------
__global__ __launch_bounds__(128)
void attn_kernel(const _Float16* __restrict__ qb,
                 const _Float16* __restrict__ kb,
                 const _Float16* __restrict__ vb,
                 const int* __restrict__ mlab,
                 _Float16* __restrict__ attn) {
  __shared__ __align__(16) _Float16 sK[32 * 64];   // [key][dim]
  __shared__ __align__(16) _Float16 sVt[64 * 32];  // [dim][key]
  __shared__ __align__(16) _Float16 sP[4 * 16 * 32];

  const int mg = blockIdx.x & 31;        // 64-row query group
  const int bh = blockIdx.x >> 5;
  const int h  = bh & 7;
  const int b  = bh >> 3;
  const int wv = threadIdx.x >> 5;
  const int lane = threadIdx.x & 31;
  const int ln = lane & 15;
  const int hi = lane >> 4;

  int pres[4];
#pragma unroll
  for (int m = 0; m < 4; ++m) pres[m] = (mlab[b * 4 + m] != 0);
  const int qok = pres[mg >> 3];         // uniform per block

  const size_t bhbase = ((size_t)(b * H_ + h)) * N_ * DH_;
  const int qrow0 = mg * 64 + wv * 16;

  // Q fragments (dims 0-31 / 32-63), pre-scaled by Dh^-0.5 = 0.125.
  const _Float16* Qg = qb + bhbase + (size_t)qrow0 * DH_;
  v16h qa0 = ldfragAB(Qg + 0, DH_);
  v16h qa1 = ldfragAB(Qg + 32, DH_);
#pragma unroll
  for (int i = 0; i < 16; ++i) {
    qa0[i] *= (_Float16)0.125f;
    qa1[i] *= (_Float16)0.125f;
  }

  float mrow[8], lrow[8];
#pragma unroll
  for (int r = 0; r < 8; ++r) { mrow[r] = -3.0e38f; lrow[r] = 0.0f; }
  v8f accv[4] = {v8f{}, v8f{}, v8f{}, v8f{}};

  _Float16* Pl = &sP[wv * 16 * 32];

  for (int kt = 0; kt < N_ / 32; ++kt) {
    const int kok = pres[(kt * 32) >> 9];
    if (!(qok & kok)) continue;          // whole tile masked (block-uniform)

    const _Float16* Kg = kb + bhbase + (size_t)(kt * 32) * DH_;
    const _Float16* Vg = vb + bhbase + (size_t)(kt * 32) * DH_;
    if (kt + 1 < N_ / 32)
      __builtin_prefetch(Kg + 32 * DH_, 0, 0);   // global_prefetch_b8

    // Stage: K via async DMA (memory->LDS, no VGPR bounce); V via registers
    // with transpose. 256 chunks of 16B, 128 threads -> 2 each.
#pragma unroll
    for (int i = 0; i < 2; ++i) {
      const int c   = threadIdx.x + i * 128;
      const int key = c >> 3;
      const int dof = (c & 7) * 8;
      async_b128(&sK[key * 64 + dof], Kg + key * 64 + dof);
      v8h vv = *(const v8h*)(Vg + key * 64 + dof);
#pragma unroll
      for (int j = 0; j < 8; ++j) sVt[(dof + j) * 32 + key] = vv[j];
    }
    wait_async0();
    __syncthreads();

    // S = Q K^T : 16x32 scores as two 16x16 C-fragments.
    v8f s0 = v8f{}, s1 = v8f{};
    {
      v16h k0a = ldfragAB(&sK[0 * 64 + 0], 64);
      v16h k0b = ldfragAB(&sK[0 * 64 + 32], 64);
      v16h k1a = ldfragAB(&sK[16 * 64 + 0], 64);
      v16h k1b = ldfragAB(&sK[16 * 64 + 32], 64);
      s0 = wmma_f16(qa0, k0a, s0);
      s0 = wmma_f16(qa1, k0b, s0);
      s1 = wmma_f16(qa0, k1a, s1);
      s1 = wmma_f16(qa1, k1b, s1);
    }

    // Online softmax; rows live in 16-lane groups, xor-shuffle reductions.
    float rmax[8], rsum[8], alpha[8];
#pragma unroll
    for (int r = 0; r < 8; ++r) rmax[r] = fmaxf(s0[r], s1[r]);
#pragma unroll
    for (int m = 1; m <= 8; m <<= 1)
#pragma unroll
      for (int r = 0; r < 8; ++r)
        rmax[r] = fmaxf(rmax[r], __shfl_xor(rmax[r], m, 32));

#pragma unroll
    for (int r = 0; r < 8; ++r) {
      const float mn = fmaxf(mrow[r], rmax[r]);
      alpha[r] = __expf(mrow[r] - mn);
      const float p0 = __expf(s0[r] - mn);
      const float p1 = __expf(s1[r] - mn);
      const int prow = hi * 8 + r;
      Pl[prow * 32 + ln]      = (_Float16)p0;
      Pl[prow * 32 + 16 + ln] = (_Float16)p1;
      rsum[r] = p0 + p1;
      mrow[r] = mn;
    }
#pragma unroll
    for (int m = 1; m <= 8; m <<= 1)
#pragma unroll
      for (int r = 0; r < 8; ++r) rsum[r] += __shfl_xor(rsum[r], m, 32);
#pragma unroll
    for (int r = 0; r < 8; ++r) lrow[r] = lrow[r] * alpha[r] + rsum[r];

#pragma unroll
    for (int f = 0; f < 4; ++f)
#pragma unroll
      for (int r = 0; r < 8; ++r) accv[f][r] *= alpha[r];

    // O += P V : P reread from LDS in A-layout; Vt gives contiguous B-frags.
    v16h pa = ldfragAB(Pl, 32);
#pragma unroll
    for (int f = 0; f < 4; ++f) {
      v16h vbf = ldfragAB(&sVt[(f * 16) * 32], 32);
      accv[f] = wmma_f16(pa, vbf, accv[f]);
    }
    __syncthreads();
  }

  float inv[8];
#pragma unroll
  for (int r = 0; r < 8; ++r) inv[r] = (lrow[r] > 0.0f) ? 1.0f / lrow[r] : 0.0f;

#pragma unroll
  for (int f = 0; f < 4; ++f)
#pragma unroll
    for (int r = 0; r < 8; ++r) {
      const int n = qrow0 + hi * 8 + r;
      attn[((size_t)(b * N_ + n)) * DIM_ + h * 64 + f * 16 + ln] =
          (_Float16)(accv[f][r] * inv[r]);
    }
}

// ---------------------------------------------------------------------------
// Output projection: same block-tiled GEMM (256x64/block, 32 rows/wave);
// +bias, f32 out.
// ---------------------------------------------------------------------------
__global__ __launch_bounds__(256)
void out_gemm_kernel(const _Float16* __restrict__ attn,
                     const _Float16* __restrict__ wT,
                     const float* __restrict__ bout,
                     float* __restrict__ out) {
  __shared__ __align__(16) _Float16 sB[2][64 * 32];

  const int NB = DIM_ / 64;                   // 8 col-blocks
  const int bm = blockIdx.x / NB;
  const int bn = blockIdx.x % NB;
  const int wv = threadIdx.x >> 5;
  const int row0 = bm * 256 + wv * 32;
  const int col0 = bn * 64;

  const int lane = threadIdx.x & 31;
  const int ln   = lane & 15;
  const int hi   = lane >> 4;

  const int scol = threadIdx.x >> 2;
  const int sko  = (threadIdx.x & 3) * 8;
  const _Float16* bsrc = wT + (size_t)(col0 + scol) * DIM_ + sko;

  async_b128(&sB[0][scol * 32 + sko], bsrc);
  wait_async0();
  __syncthreads();

  const _Float16* arow = attn + (size_t)row0 * DIM_;
  v8f acc[2][4] = {{v8f{}, v8f{}, v8f{}, v8f{}}, {v8f{}, v8f{}, v8f{}, v8f{}}};
  v16h a0 = ldfragAB(arow, DIM_);
  v16h a1 = ldfragAB(arow + (size_t)16 * DIM_, DIM_);

  int buf = 0;
  for (int kk = 0; kk < DIM_; kk += 32, buf ^= 1) {
    if (kk + 32 < DIM_)
      async_b128(&sB[buf ^ 1][scol * 32 + sko], bsrc + kk + 32);

    v16h a0n = a0, a1n = a1;
    if (kk + 32 < DIM_) {
      a0n = ldfragAB(arow + kk + 32, DIM_);
      a1n = ldfragAB(arow + (size_t)16 * DIM_ + kk + 32, DIM_);
    }

#pragma unroll
    for (int j = 0; j < 4; ++j) {
      v16h bfr = ldfragAB(&sB[buf][(j * 16) * 32], 32);
      acc[0][j] = wmma_f16(a0, bfr, acc[0][j]);
      acc[1][j] = wmma_f16(a1, bfr, acc[1][j]);
    }
    a0 = a0n;
    a1 = a1n;
    wait_async0();
    __syncthreads();
  }

#pragma unroll
  for (int m = 0; m < 2; ++m) {
#pragma unroll
    for (int j = 0; j < 4; ++j) {
#pragma unroll
      for (int r = 0; r < 8; ++r) {
        const int row = row0 + m * 16 + hi * 8 + r;
        const int col = col0 + j * 16 + ln;
        out[(size_t)row * DIM_ + col] = acc[m][j][r] + bout[col];
      }
    }
  }
}

// ---------------------------------------------------------------------------
// Workspace layout (bytes)
// ---------------------------------------------------------------------------
static constexpr size_t OFF_XN    = 0;
static constexpr size_t OFF_WQKVT = OFF_XN    + (size_t)ROWS_ * DIM_ * 2;
static constexpr size_t OFF_WOUTT = OFF_WQKVT + (size_t)QKVC_ * DIM_ * 2;
static constexpr size_t OFF_Q     = OFF_WOUTT + (size_t)DIM_ * DIM_ * 2;
static constexpr size_t OFF_K     = OFF_Q     + (size_t)ROWS_ * DIM_ * 2;
static constexpr size_t OFF_V     = OFF_K     + (size_t)ROWS_ * DIM_ * 2;
static constexpr size_t OFF_AT    = OFF_V     + (size_t)ROWS_ * DIM_ * 2;

extern "C" void kernel_launch(void* const* d_in, const int* in_sizes, int n_in,
                              void* d_out, int out_size, void* d_ws, size_t ws_size,
                              hipStream_t stream) {
  const float* x      = (const float*)d_in[0];
  const int*   mlab   = (const int*)d_in[1];
  const float* gamma  = (const float*)d_in[2];
  const float* beta   = (const float*)d_in[3];
  const float* w_qkv  = (const float*)d_in[4];
  const float* w_out  = (const float*)d_in[5];
  const float* b_out  = (const float*)d_in[6];
  float* out = (float*)d_out;

  char* ws = (char*)d_ws;
  _Float16* xn    = (_Float16*)(ws + OFF_XN);
  _Float16* wqkvT = (_Float16*)(ws + OFF_WQKVT);
  _Float16* woutT = (_Float16*)(ws + OFF_WOUTT);
  _Float16* qb    = (_Float16*)(ws + OFF_Q);
  _Float16* kb    = (_Float16*)(ws + OFF_K);
  _Float16* vb    = (_Float16*)(ws + OFF_V);
  _Float16* at    = (_Float16*)(ws + OFF_AT);

  cvt_wqkv_kernel<<<(DIM_ * QKVC_ + 255) / 256, 256, 0, stream>>>(w_qkv, wqkvT);
  cvt_wout_kernel<<<(DIM_ * DIM_ + 255) / 256, 256, 0, stream>>>(w_out, woutT);
  ln_kernel<<<ROWS_, 256, 0, stream>>>(x, gamma, beta, xn);
  qkv_gemm_kernel<<<(ROWS_ / 256) * (QKVC_ / 64), 256, 0, stream>>>(
      xn, wqkvT, qb, kb, vb);
  attn_kernel<<<B_ * H_ * (N_ / 64), 128, 0, stream>>>(qb, kb, vb, mlab, at);
  out_gemm_kernel<<<(ROWS_ / 256) * (DIM_ / 64), 256, 0, stream>>>(
      at, woutT, b_out, out);
}
